// GATClassifier_13486197310302
// MI455X (gfx1250) — compile-verified, matrix operand order checked by reading, not run on hardware
//
#include <hip/hip_runtime.h>
#include <cstdint>
#include <cstddef>

#define N_NODES 50000
#define N_EDGES 800000
#define IN_DIM  256
#define HID     32
#define HEADS   8
#define OUT_DIM 16
#define E_TOT   (N_EDGES + N_NODES)   // self-loops appended

typedef __attribute__((ext_vector_type(16))) _Float16 v16h;
typedef __attribute__((ext_vector_type(8)))  _Float16 h8v;   // one 16B chunk
typedef __attribute__((ext_vector_type(8)))  float    v8f;

// ---------------------------------------------------------------- utilities

__global__ void k_fill_u32(uint32_t* __restrict__ p, uint32_t v, long long n) {
  long long i = blockIdx.x * (long long)blockDim.x + threadIdx.x;
  if (i < n) p[i] = v;
}

__global__ void k_f32_to_f16(const float* __restrict__ in, _Float16* __restrict__ out,
                             long long n) {
  long long i = blockIdx.x * (long long)blockDim.x + threadIdx.x;
  if (i < n) out[i] = (_Float16)in[i];
}

// in: [R,C] row-major f32  ->  out: [C,R] f16 (column-major view of in)
__global__ void k_transpose_f16(const float* __restrict__ in, _Float16* __restrict__ out,
                                int R, int C) {
  int t = blockIdx.x * blockDim.x + threadIdx.x;
  if (t >= R * C) return;
  int r = t / C, c = t % C;
  out[(size_t)c * R + r] = (_Float16)in[(size_t)r * C + c];
}

__device__ __forceinline__ void atomicMaxFloat(float* addr, float val) {
  // int view is monotone for non-negative floats, uint view inversely for negative
  if (val >= 0.0f) atomicMax((int*)addr, __float_as_int(val));
  else             atomicMin((unsigned int*)addr, __float_as_uint(val));
}

__device__ __forceinline__ float lrelu(float a) { return a > 0.0f ? a : 0.2f * a; }

// ---------------------------------------------------------------- WMMA GEMM
// C[M,Nc] = A[M,256] * B[256,Nc]; A row-major f16, Bt = B^T ([Nc,256]) f16.
// One wave per 16-row tile x (CT*16)-col tile; 8 waves/block along M.
// Fragment layout per CDNA5 ISA 7.12.2: lane = M (or N) mod 16; lane/16
// selects the +8 K half; each lane's 16 halves are two contiguous 16B runs
// at K = kk+grp*8 and K = kk+16+grp*8  ->  two global_load_b128 per fragment.
template <int CT>
__global__ void k_gemm_wmma(const _Float16* __restrict__ A,
                            const _Float16* __restrict__ Bt,
                            float* __restrict__ C,
                            int M, int Nc) {
  constexpr int K = 256;
  const int lane    = threadIdx.x & 31;
  const int wave    = threadIdx.x >> 5;
  const int rowTile = blockIdx.y * 8 + wave;
  const int row0    = rowTile * 16;
  const int col0    = blockIdx.x * (16 * CT);
  if (row0 >= M) return;                     // wave-uniform exit, EXEC stays full
  const int grp = lane >> 4;                 // selects K+0 / K+8 half per chunk
  const int lr  = lane & 15;
  const int arow = min(row0 + lr, M - 1);    // clamp: row r only feeds C row r,
                                             // rows >= M are never stored
  const _Float16* ap = A + (size_t)arow * K + grp * 8;
  const _Float16* bp[CT];
#pragma unroll
  for (int j = 0; j < CT; ++j)
    bp[j] = Bt + (size_t)(col0 + j * 16 + lr) * K + grp * 8;

  v8f acc[CT];
#pragma unroll
  for (int j = 0; j < CT; ++j) acc[j] = (v8f){};

#pragma unroll
  for (int kk = 0; kk < K; kk += 32) {
    h8v a0 = *(const h8v*)(ap + kk);
    h8v a1 = *(const h8v*)(ap + kk + 16);
    v16h av = __builtin_shufflevector(a0, a1, 0, 1, 2, 3, 4, 5, 6, 7,
                                      8, 9, 10, 11, 12, 13, 14, 15);
#pragma unroll
    for (int j = 0; j < CT; ++j) {
      h8v b0 = *(const h8v*)(bp[j] + kk);
      h8v b1 = *(const h8v*)(bp[j] + kk + 16);
      v16h bv = __builtin_shufflevector(b0, b1, 0, 1, 2, 3, 4, 5, 6, 7,
                                        8, 9, 10, 11, 12, 13, 14, 15);
      acc[j] = __builtin_amdgcn_wmma_f32_16x16x32_f16(false, av, false, bv,
                                                      (short)0, acc[j],
                                                      false, false);
    }
  }
#pragma unroll
  for (int r = 0; r < 8; ++r) {
    const int m = row0 + grp * 8 + r;        // VGPR r -> M=r (lanes 0-15) / 8+r
    if (m < M) {
#pragma unroll
      for (int j = 0; j < CT; ++j)
        C[(size_t)m * Nc + col0 + j * 16 + lr] = acc[j][r];
    }
  }
}

// ---------------------------------------------------------------- attention

// a_s[n,h] = sum_c H[n,h,c]*att_s[h,c]; same for a_d. Thread per (n,h).
template <int H, int C>
__global__ void k_attn(const float* __restrict__ Hm,
                       const float* __restrict__ atts, const float* __restrict__ attd,
                       float* __restrict__ as, float* __restrict__ ad) {
  long long t = blockIdx.x * (long long)blockDim.x + threadIdx.x;
  if (t >= (long long)N_NODES * H) return;
  const int n = (int)(t / H), h = (int)(t % H);
  const float* hp = Hm + ((size_t)n * H + h) * C;
  float ss = 0.0f, dd = 0.0f;
#pragma unroll
  for (int c = 0; c < C; ++c) { float v = hp[c]; ss += v * atts[h * C + c]; dd += v * attd[h * C + c]; }
  as[t] = ss; ad[t] = dd;
}

__device__ __forceinline__ void edge_sd(int e, const int* esrc, const int* edst,
                                        int& s, int& d) {
  if (e < N_EDGES) { s = esrc[e]; d = edst[e]; }
  else             { s = e - N_EDGES; d = s; }      // self-loop
}

// pass 1: per-(edge,head) segment max of leaky_relu(a_s[src]+a_d[dst])
template <int H>
__global__ void k_edge_max(const int* __restrict__ esrc, const int* __restrict__ edst,
                           const float* __restrict__ as, const float* __restrict__ ad,
                           float* __restrict__ m) {
  long long t = blockIdx.x * (long long)blockDim.x + threadIdx.x;
  if (t >= (long long)E_TOT * H) return;
  const int e = (int)(t / H), h = (int)(t % H);
  int s, d; edge_sd(e, esrc, edst, s, d);
  atomicMaxFloat(&m[d * H + h], lrelu(as[s * H + h] + ad[d * H + h]));
}

// pass 2: per-(edge,head) sum of exp(alpha - max)
template <int H>
__global__ void k_edge_sum(const int* __restrict__ esrc, const int* __restrict__ edst,
                           const float* __restrict__ as, const float* __restrict__ ad,
                           const float* __restrict__ m, float* __restrict__ ssum) {
  long long t = blockIdx.x * (long long)blockDim.x + threadIdx.x;
  if (t >= (long long)E_TOT * H) return;
  const int e = (int)(t / H), h = (int)(t % H);
  int s, d; edge_sd(e, esrc, edst, s, d);
  float a = lrelu(as[s * H + h] + ad[d * H + h]);
  atomicAdd(&ssum[d * H + h], __expf(a - m[d * H + h]));
}

// pass 3 layer1: one block per edge, 256 channels; coef per head via LDS
__global__ void k_edge_agg1(const int* __restrict__ esrc, const int* __restrict__ edst,
                            const float* __restrict__ as, const float* __restrict__ ad,
                            const float* __restrict__ m, const float* __restrict__ ssum,
                            const float* __restrict__ Hm, float* __restrict__ outacc) {
  __shared__ float coef[HEADS];
  const int e = blockIdx.x;
  int s, d; edge_sd(e, esrc, edst, s, d);
  if (threadIdx.x < HEADS) {
    const int h = threadIdx.x;
    float a = lrelu(as[s * HEADS + h] + ad[d * HEADS + h]);
    coef[h] = __expf(a - m[d * HEADS + h]) / (ssum[d * HEADS + h] + 1e-16f);
  }
  __syncthreads();
  const int c = threadIdx.x;                       // 0..255, head = c>>5
  atomicAdd(&outacc[(size_t)d * 256 + c], coef[c >> 5] * Hm[(size_t)s * 256 + c]);
}

// pass 3 layer2: 16 channels, 1 head; 16 edges per block
__global__ void k_edge_agg2(const int* __restrict__ esrc, const int* __restrict__ edst,
                            const float* __restrict__ as, const float* __restrict__ ad,
                            const float* __restrict__ m, const float* __restrict__ ssum,
                            const float* __restrict__ h2, float* __restrict__ out) {
  long long t = blockIdx.x * 256LL + threadIdx.x;
  const int e = (int)(t >> 4);
  if (e >= E_TOT) return;
  const int c = (int)(t & 15);
  int s, d; edge_sd(e, esrc, edst, s, d);
  float a = lrelu(as[s] + ad[d]);
  const float coef = __expf(a - m[d]) / (ssum[d] + 1e-16f);
  atomicAdd(&out[(size_t)d * 16 + c], coef * h2[(size_t)s * 16 + c]);
}

// bias + ELU + f16 convert (layer-1 epilogue feeding GEMM2)
__global__ void k_post1(const float* __restrict__ acc, const float* __restrict__ b1,
                        _Float16* __restrict__ outh) {
  long long t = blockIdx.x * (long long)blockDim.x + threadIdx.x;
  if (t >= (long long)N_NODES * 256) return;
  float v = acc[t] + b1[(int)(t & 255)];
  v = v > 0.0f ? v : __expf(v) - 1.0f;             // ELU
  outh[t] = (_Float16)v;
}

__global__ void k_bias2(float* __restrict__ out, const float* __restrict__ b2) {
  long long t = blockIdx.x * (long long)blockDim.x + threadIdx.x;
  if (t >= (long long)N_NODES * 16) return;
  out[t] += b2[(int)(t & 15)];
}

// ---------------------------------------------------------------- launch

extern "C" void kernel_launch(void* const* d_in, const int* in_sizes, int n_in,
                              void* d_out, int out_size, void* d_ws, size_t ws_size,
                              hipStream_t stream) {
  const float* x    = (const float*)d_in[0];
  const int*   ei   = (const int*)d_in[1];
  const int*   esrc = ei;
  const int*   edst = ei + N_EDGES;
  const float* W1   = (const float*)d_in[2];
  const float* at_s1= (const float*)d_in[3];
  const float* at_d1= (const float*)d_in[4];
  const float* b1   = (const float*)d_in[5];
  const float* W2   = (const float*)d_in[6];
  const float* at_s2= (const float*)d_in[7];
  const float* at_d2= (const float*)d_in[8];
  const float* b2   = (const float*)d_in[9];
  float* out = (float*)d_out;

  char* w = (char*)d_ws;
  auto carve = [&](size_t bytes) -> char* {
    char* p = w; w += (bytes + 255) & ~(size_t)255; return p;
  };
  _Float16* xh  = (_Float16*)carve((size_t)N_NODES * 256 * 2);
  _Float16* W1t = (_Float16*)carve((size_t)256 * 256 * 2);     // [n][k]
  _Float16* W2t = (_Float16*)carve((size_t)16 * 256 * 2);      // [n][k]
  float* H1     = (float*)carve((size_t)N_NODES * 256 * 4);    // x@W1
  float* out1   = (float*)carve((size_t)N_NODES * 256 * 4);    // aggregated
  _Float16* h1h = (_Float16*)carve((size_t)N_NODES * 256 * 2); // elu(out1+b1) f16
  float* a_s1   = (float*)carve((size_t)N_NODES * 8 * 4);
  float* a_d1   = (float*)carve((size_t)N_NODES * 8 * 4);
  float* m1     = (float*)carve((size_t)N_NODES * 8 * 4);
  float* s1     = (float*)carve((size_t)N_NODES * 8 * 4);
  float* h2     = (float*)carve((size_t)N_NODES * 16 * 4);     // h1@W2
  float* a_s2   = (float*)carve((size_t)N_NODES * 4);
  float* a_d2   = (float*)carve((size_t)N_NODES * 4);
  float* m2     = (float*)carve((size_t)N_NODES * 4);
  float* s2     = (float*)carve((size_t)N_NODES * 4);

  auto nb = [](long long n) { return (unsigned)((n + 255) / 256); };
  const uint32_t NEG_INF = 0xFF800000u;

  // init accumulators (deterministic every call)
  k_fill_u32<<<nb((long long)N_NODES*8), 256, 0, stream>>>((uint32_t*)m1, NEG_INF, (long long)N_NODES*8);
  k_fill_u32<<<nb((long long)N_NODES*8), 256, 0, stream>>>((uint32_t*)s1, 0u, (long long)N_NODES*8);
  k_fill_u32<<<nb((long long)N_NODES*256), 256, 0, stream>>>((uint32_t*)out1, 0u, (long long)N_NODES*256);
  k_fill_u32<<<nb(N_NODES), 256, 0, stream>>>((uint32_t*)m2, NEG_INF, N_NODES);
  k_fill_u32<<<nb(N_NODES), 256, 0, stream>>>((uint32_t*)s2, 0u, N_NODES);
  k_fill_u32<<<nb((long long)N_NODES*16), 256, 0, stream>>>((uint32_t*)out, 0u, (long long)N_NODES*16);

  // precision staging for WMMA
  k_f32_to_f16<<<nb((long long)N_NODES*256), 256, 0, stream>>>(x, xh, (long long)N_NODES*256);
  k_transpose_f16<<<nb(256*256), 256, 0, stream>>>(W1, W1t, 256, 256);
  k_transpose_f16<<<nb(256*16), 256, 0, stream>>>(W2, W2t, 256, 16);

  // ---- layer 1 ----
  {
    dim3 grid(256 / (16 * 4), (N_NODES / 16 + 7) / 8);   // 4 x 391
    k_gemm_wmma<4><<<grid, 256, 0, stream>>>(xh, W1t, H1, N_NODES, 256);
  }
  k_attn<HEADS, HID><<<nb((long long)N_NODES*8), 256, 0, stream>>>(H1, at_s1, at_d1, a_s1, a_d1);
  k_edge_max<HEADS><<<nb((long long)E_TOT*8), 256, 0, stream>>>(esrc, edst, a_s1, a_d1, m1);
  k_edge_sum<HEADS><<<nb((long long)E_TOT*8), 256, 0, stream>>>(esrc, edst, a_s1, a_d1, m1, s1);
  k_edge_agg1<<<E_TOT, 256, 0, stream>>>(esrc, edst, a_s1, a_d1, m1, s1, H1, out1);
  k_post1<<<nb((long long)N_NODES*256), 256, 0, stream>>>(out1, b1, h1h);

  // ---- layer 2 ----
  {
    dim3 grid(1, (N_NODES / 16 + 7) / 8);                // 1 x 391
    k_gemm_wmma<1><<<grid, 256, 0, stream>>>(h1h, W2t, h2, N_NODES, 16);
  }
  k_attn<1, OUT_DIM><<<nb((long long)N_NODES), 256, 0, stream>>>(h2, at_s2, at_d2, a_s2, a_d2);
  k_edge_max<1><<<nb((long long)E_TOT), 256, 0, stream>>>(esrc, edst, a_s2, a_d2, m2);
  k_edge_sum<1><<<nb((long long)E_TOT), 256, 0, stream>>>(esrc, edst, a_s2, a_d2, m2, s2);
  k_edge_agg2<<<nb((long long)E_TOT*16), 256, 0, stream>>>(esrc, edst, a_s2, a_d2, m2, s2, h2, out);
  k_bias2<<<nb((long long)N_NODES*16), 256, 0, stream>>>(out, b2);
}
